// MambaBlock_85796266705208
// MI455X (gfx1250) — matrix operand order repaired and checked by reading, not compiled, not run on hardware
//
#include <hip/hip_runtime.h>
#include <hip/hip_bf16.h>

// ---------------- problem constants ----------------
#define B_SZ   2
#define L_SZ   2048
#define DM_SZ  1024
#define DS_SZ  16
#define DC_SZ  4
#define DI_SZ  2048            // EXP * DM
#define DTR_SZ 64              // DM / 16
#define ROWS   (B_SZ * L_SZ)   // 4096

typedef unsigned short u16;
typedef __attribute__((ext_vector_type(16))) __bf16 v16bf;
typedef __attribute__((ext_vector_type(8)))  float  v8f;

struct U4 { unsigned int a, b, c, d; };   // POD 16-byte chunk (8 bf16)

__device__ __forceinline__ u16 f32_to_bf16(float f) {
    unsigned int u = __float_as_uint(f);
    unsigned int r = u + 0x7FFFu + ((u >> 16) & 1u);   // round-to-nearest-even
    return (u16)(r >> 16);
}

__device__ __forceinline__ float silu_f(float x) {
    return x / (1.0f + __expf(-x));
}

// ---------------- elementwise fp32 -> bf16 ----------------
__global__ void cvt_f32_bf16(const float* __restrict__ in, u16* __restrict__ out, size_t n) {
    size_t i = (size_t)blockIdx.x * blockDim.x + threadIdx.x;
    size_t stride = (size_t)gridDim.x * blockDim.x;
    for (; i < n; i += stride) out[i] = f32_to_bf16(in[i]);
}

// C/D 16x16 f32 layout: VGPR i, lanes 0-15 -> M=i, lanes 16-31 -> M=i+8; N = lane&15.
__device__ __forceinline__ void store_c_tile(float* __restrict__ C, int ldc, int m0, int n0, v8f acc) {
    int lane = threadIdx.x & 31;
    int n = n0 + (lane & 15);
    int mb = m0 + ((lane >> 4) << 3);
#pragma unroll
    for (int i = 0; i < 8; ++i) C[(size_t)(mb + i) * ldc + n] = acc[i];
}

// ---------------- generic NT GEMM: C[MxN] = A[MxK] * B[NxK]^T ----------------
// One wave computes a (16*MT)x(16*NT) tile; all tile loops compile-time unrolled
// (fixed-VGPR accumulators, no movrel). 8 waves / block.
//
// A-frag (16x32 bf16, ISA 7.12.2): lane m = lane&15;
//   lanes 0-15 hold K {k0..k0+7, k0+16..k0+23}; lanes 16-31 hold K {+8..+15, +24..+31}.
// B-frag (32x16, B stored row-major [N][K]): lane n = lane&15;
//   lanes 0-15 hold K {k0..k0+7, k0+16..k0+23}; lanes 16-31 hold K {+8..+15, +24..+31} (same as A).
template<int MT, int NT>
__global__ __launch_bounds__(256) void gemm_bf16_nt(const u16* __restrict__ A,
                                                    const u16* __restrict__ Bm,
                                                    float* __restrict__ C,
                                                    int M, int N, int K) {
    int wave = blockIdx.x * (blockDim.x >> 5) + (threadIdx.x >> 5);
    int nGroups = N / (16 * NT);
    int mGroups = M / (16 * MT);
    if (wave >= mGroups * nGroups) return;          // wave-uniform
    int mg = wave / nGroups;
    int ng = wave % nGroups;
    int m0 = mg * 16 * MT;
    int n0 = ng * 16 * NT;
    int lane = threadIdx.x & 31;

    // hoisted per-lane row bases (K-loop reduces to constant byte increments)
    const u16* aRow[MT];
#pragma unroll
    for (int i = 0; i < MT; ++i)
        aRow[i] = A + (size_t)(m0 + i * 16 + (lane & 15)) * K + ((lane >> 4) << 3);
    const u16* bRow[NT];
#pragma unroll
    for (int j = 0; j < NT; ++j)
        bRow[j] = Bm + (size_t)(n0 + j * 16 + (lane & 15)) * K + ((lane >> 4) << 3);

    v8f acc[MT][NT] = {};

#pragma unroll 2
    for (int k0 = 0; k0 < K; k0 += 32) {
        union { v16bf v; U4 q[2]; } a[MT], b[NT];
#pragma unroll
        for (int i = 0; i < MT; ++i) {
            a[i].q[0] = *(const U4*)(aRow[i] + k0);        // K = k0 .. +7
            a[i].q[1] = *(const U4*)(aRow[i] + k0 + 16);   // K = k0+16 .. +23
        }
#pragma unroll
        for (int j = 0; j < NT; ++j) {
            b[j].q[0] = *(const U4*)(bRow[j] + k0);
            b[j].q[1] = *(const U4*)(bRow[j] + k0 + 16);
        }
#pragma unroll
        for (int i = 0; i < MT; ++i)
#pragma unroll
            for (int j = 0; j < NT; ++j)
                acc[i][j] = __builtin_amdgcn_wmma_f32_16x16x32_bf16(
                    false, a[i].v, false, b[j].v, (short)0, acc[i][j], false, false);
    }

#pragma unroll
    for (int i = 0; i < MT; ++i)
#pragma unroll
        for (int j = 0; j < NT; ++j)
            store_c_tile(C, N, m0 + i * 16, n0 + j * 16, acc[i][j]);
}

// ---------------- depthwise causal conv (DC=4) + bias + SiLU ----------------
__global__ void conv_silu_k(const float* __restrict__ xz,
                            const float* __restrict__ Wc,   // [DI][DC]
                            const float* __restrict__ bc,   // [DI]
                            float* __restrict__ u, u16* __restrict__ u_bf) {
    size_t i = (size_t)blockIdx.x * blockDim.x + threadIdx.x;
    const size_t total = (size_t)ROWS * DI_SZ;
    if (i >= total) return;
    int d  = (int)(i % DI_SZ);
    size_t bl = i / DI_SZ;
    int t  = (int)(bl % L_SZ);
    size_t brow = bl - t;                 // b*L
    float acc = bc[d];
#pragma unroll
    for (int j = 0; j < DC_SZ; ++j) {
        int tt = t - (DC_SZ - 1) + j;
        if (tt >= 0)
            acc += xz[(brow + tt) * (size_t)(2 * DI_SZ) + d] * Wc[d * DC_SZ + j];
    }
    float s = silu_f(acc);
    u[i] = s;
    u_bf[i] = f32_to_bf16(s);
}

// ---------------- slice proj[:, :DTR] -> bf16 dtr ----------------
__global__ void slice_dtr_k(const float* __restrict__ proj, u16* __restrict__ dtr_bf) {
    size_t i = (size_t)blockIdx.x * blockDim.x + threadIdx.x;
    const size_t total = (size_t)ROWS * DTR_SZ;
    if (i >= total) return;
    size_t r = i / DTR_SZ;
    int c = (int)(i % DTR_SZ);
    dtr_bf[i] = f32_to_bf16(proj[r * (DTR_SZ + 2 * DS_SZ) + c]);
}

// ---------------- selective scan + D-skip + SiLU(z) gate -> bf16 ----------------
__global__ __launch_bounds__(256) void scan_k(const float* __restrict__ draw,   // [ROWS][DI] pre-softplus
                                              const float* __restrict__ b_dt,   // [DI]
                                              const float* __restrict__ A_log,  // [DI][DS]
                                              const float* __restrict__ u,      // [ROWS][DI]
                                              const float* __restrict__ proj,   // [ROWS][96]
                                              const float* __restrict__ xz,     // [ROWS][2*DI]
                                              const float* __restrict__ D_skip, // [DI]
                                              u16* __restrict__ gated_bf) {     // [ROWS][DI]
    const int blocksPerB = DI_SZ / 256;
    int b    = blockIdx.x / blocksPerB;
    int d    = (blockIdx.x % blocksPerB) * 256 + threadIdx.x;
    const int LDP = DTR_SZ + 2 * DS_SZ;   // 96

    float A[DS_SZ], h[DS_SZ];
#pragma unroll
    for (int n = 0; n < DS_SZ; ++n) { A[n] = -__expf(A_log[d * DS_SZ + n]); h[n] = 0.0f; }
    float bd = b_dt[d];
    float Dk = D_skip[d];

    __shared__ float sBC[2 * DS_SZ];      // B_t (16) then C_t (16)

    for (int t = 0; t < L_SZ; ++t) {
        size_t row = (size_t)b * L_SZ + t;
        if (threadIdx.x < 2 * DS_SZ)
            sBC[threadIdx.x] = proj[row * LDP + DTR_SZ + threadIdx.x];
        __syncthreads();

        float x = draw[row * DI_SZ + d] + bd;
        float dlt = (x > 20.0f) ? x : log1pf(__expf(x));   // softplus
        float ut  = u[row * DI_SZ + d];
        float y = 0.0f;
#pragma unroll
        for (int n = 0; n < DS_SZ; ++n) {
            float dA = __expf(dlt * A[n]);
            h[n] = dA * h[n] + dlt * sBC[n] * ut;
            y += h[n] * sBC[DS_SZ + n];
        }
        float z = xz[row * (size_t)(2 * DI_SZ) + DI_SZ + d];
        float g = (y + ut * Dk) * silu_f(z);
        gated_bf[row * DI_SZ + d] = f32_to_bf16(g);
        __syncthreads();                   // protect sBC before next overwrite
    }
}

// ---------------- host launcher ----------------
extern "C" void kernel_launch(void* const* d_in, const int* in_sizes, int n_in,
                              void* d_out, int out_size, void* d_ws, size_t ws_size,
                              hipStream_t stream) {
    const float* x      = (const float*)d_in[0];   // [B,L,DM]
    const float* W_in   = (const float*)d_in[1];   // [2*DI, DM]
    const float* W_conv = (const float*)d_in[2];   // [DI, DC]
    const float* b_conv = (const float*)d_in[3];   // [DI]
    const float* W_xp   = (const float*)d_in[4];   // [96, DI]
    const float* W_dt   = (const float*)d_in[5];   // [DI, DTR]
    const float* b_dt   = (const float*)d_in[6];   // [DI]
    const float* A_log  = (const float*)d_in[7];   // [DI, DS]
    const float* D_skip = (const float*)d_in[8];   // [DI]
    const float* W_out  = (const float*)d_in[9];   // [DM, DI]
    float* out = (float*)d_out;                    // [B,L,DM] fp32

    // ---- carve workspace ----
    char* p = (char*)d_ws;
    auto take = [&](size_t bytes) { char* r = p; p += (bytes + 255) & ~(size_t)255; return r; };
    u16*   x_bf    = (u16*)  take((size_t)ROWS * DM_SZ * 2);
    u16*   Win_bf  = (u16*)  take((size_t)2 * DI_SZ * DM_SZ * 2);
    u16*   Wout_bf = (u16*)  take((size_t)DM_SZ * DI_SZ * 2);
    u16*   Wxp_bf  = (u16*)  take((size_t)96 * DI_SZ * 2);
    u16*   Wdt_bf  = (u16*)  take((size_t)DI_SZ * DTR_SZ * 2);
    float* xz      = (float*)take((size_t)ROWS * 2 * DI_SZ * 4);
    float* u_f     = (float*)take((size_t)ROWS * DI_SZ * 4);
    u16*   u_bf    = (u16*)  take((size_t)ROWS * DI_SZ * 2);
    float* proj    = (float*)take((size_t)ROWS * 96 * 4);
    u16*   dtr_bf  = (u16*)  take((size_t)ROWS * DTR_SZ * 2);
    float* draw    = (float*)take((size_t)ROWS * DI_SZ * 4);
    u16*   gated   = (u16*)  take((size_t)ROWS * DI_SZ * 2);

    const int TB = 256;
    auto gs = [&](size_t n) { return (unsigned)((n + TB - 1) / TB); };

    // ---- fp32 -> bf16 conversions ----
    cvt_f32_bf16<<<gs((size_t)ROWS * DM_SZ),      TB, 0, stream>>>(x,     x_bf,    (size_t)ROWS * DM_SZ);
    cvt_f32_bf16<<<gs((size_t)2 * DI_SZ * DM_SZ), TB, 0, stream>>>(W_in,  Win_bf,  (size_t)2 * DI_SZ * DM_SZ);
    cvt_f32_bf16<<<gs((size_t)DM_SZ * DI_SZ),     TB, 0, stream>>>(W_out, Wout_bf, (size_t)DM_SZ * DI_SZ);
    cvt_f32_bf16<<<gs((size_t)96 * DI_SZ),        TB, 0, stream>>>(W_xp,  Wxp_bf,  (size_t)96 * DI_SZ);
    cvt_f32_bf16<<<gs((size_t)DI_SZ * DTR_SZ),    TB, 0, stream>>>(W_dt,  Wdt_bf,  (size_t)DI_SZ * DTR_SZ);

    auto blocks_for = [](int M, int N, int mt, int nt) {
        int jobs = (M / (16 * mt)) * (N / (16 * nt));
        return (unsigned)((jobs + 7) / 8);          // 8 waves / block
    };

    // 1) xz = x @ W_in^T   (4096 x 4096 x 1024), 32x64 per wave
    gemm_bf16_nt<2, 4><<<blocks_for(ROWS, 2 * DI_SZ, 2, 4), TB, 0, stream>>>(
        x_bf, Win_bf, xz, ROWS, 2 * DI_SZ, DM_SZ);

    // 2) depthwise causal conv + SiLU -> u
    conv_silu_k<<<gs((size_t)ROWS * DI_SZ), TB, 0, stream>>>(xz, W_conv, b_conv, u_f, u_bf);

    // 3) proj = u @ W_xproj^T   (4096 x 96 x 2048), 32x32 per wave (96 = 3*32)
    gemm_bf16_nt<2, 2><<<blocks_for(ROWS, 96, 2, 2), TB, 0, stream>>>(
        u_bf, Wxp_bf, proj, ROWS, 96, DI_SZ);

    // 4) dtr slice -> bf16 ; draw = dtr @ W_dt^T   (4096 x 2048 x 64)
    slice_dtr_k<<<gs((size_t)ROWS * DTR_SZ), TB, 0, stream>>>(proj, dtr_bf);
    gemm_bf16_nt<2, 4><<<blocks_for(ROWS, DI_SZ, 2, 4), TB, 0, stream>>>(
        dtr_bf, Wdt_bf, draw, ROWS, DI_SZ, DTR_SZ);

    // 5) selective scan + gate -> gated (bf16)
    scan_k<<<B_SZ * (DI_SZ / 256), TB, 0, stream>>>(draw, b_dt, A_log, u_f, proj, xz, D_skip, gated);

    // 6) out = gated @ W_out^T   (4096 x 1024 x 2048)
    gemm_bf16_nt<2, 4><<<blocks_for(ROWS, DM_SZ, 2, 4), TB, 0, stream>>>(
        gated, Wout_bf, out, ROWS, DM_SZ, DI_SZ);
}
